// CLLayer_39247411151275
// MI455X (gfx1250) — matrix operand order repaired and checked by reading, not compiled, not run on hardware
//
#include <hip/hip_runtime.h>
#include <hip/hip_bf16.h>

typedef __attribute__((ext_vector_type(16))) _Float16 v16h;
typedef __attribute__((ext_vector_type(8)))  float    v8f;
typedef __attribute__((ext_vector_type(4)))  int      v4i;

#define NROWS 8192
#define DDIM  1024
#define KC    64      // K-chunk staged in LDS
#define LSTR  72      // LDS row stride in halves (64 + 8 pad, keeps 16B align, kills bank alias)

#if defined(__has_builtin)
# if __has_builtin(__builtin_amdgcn_global_load_async_to_lds_b128)
#  define HAVE_ASYNC_LDS 1
# endif
#endif
#ifndef HAVE_ASYNC_LDS
# define HAVE_ASYNC_LDS 0
#endif

// 16-byte global -> LDS copy. Async (VGPR-free, ASYNCcnt-tracked) when available.
__device__ inline void cp16(const _Float16* g, _Float16* l) {
#if HAVE_ASYNC_LDS
  __builtin_amdgcn_global_load_async_to_lds_b128((v4i*)g, (v4i*)l, 0, 0);
#else
  *reinterpret_cast<uint4*>(l) = *reinterpret_cast<const uint4*>(g);
#endif
}
__device__ inline void wait_async_lds() {
#if HAVE_ASYNC_LDS
# if defined(__has_builtin) && __has_builtin(__builtin_amdgcn_s_wait_asynccnt)
  __builtin_amdgcn_s_wait_asynccnt(0);
# else
  asm volatile("s_wait_asynccnt 0" ::: "memory");
# endif
#endif
}

// ---------- fragment loaders (contiguous k order, same in A and B) ----------
// A 16x32 f16: lane L, row = L&15; e[i] = K[k0+i], k0=(L>>4)*16
__device__ inline v16h ld_frag_a(const _Float16* row, int k0) {
  union { uint4 u[2]; v16h h; } u;
  u.u[0] = *reinterpret_cast<const uint4*>(row + k0);
  u.u[1] = *reinterpret_cast<const uint4*>(row + k0 + 8);
  return u.h;
}
// B 32x16 f16 (NT: B[k][n] = Brow_n[k]): lane L, col = L&15; e[i] = Brow[ks + (L>>4)*16 + i]
__device__ inline v16h ld_frag_b(const _Float16* row, int k0) {
  union { uint4 u[2]; v16h h; } u;
  const uint4* p = reinterpret_cast<const uint4*>(row + k0);
  u.u[0] = p[0];
  u.u[1] = p[1];
  return u.h;
}

__device__ inline v8f wmma_f16(v16h a, v16h b, v8f c) {
  return __builtin_amdgcn_wmma_f32_16x16x32_f16(false, a, false, b, (short)0, c, false, false);
}

// ---------- elementwise helpers ----------
__global__ void cvt_f16_kernel(const float* __restrict__ in, _Float16* __restrict__ out, int n) {
  int i = blockIdx.x * blockDim.x + threadIdx.x;
  if (i < n) out[i] = (_Float16)in[i];
}

__global__ void zero_f32_kernel(float* __restrict__ p, int n) {
  int i = blockIdx.x * blockDim.x + threadIdx.x;
  if (i < n) p[i] = 0.0f;
}

// ---------- staging helpers (128 rows x 64 halves per buffer) ----------
__device__ inline void stage_gemm(const _Float16* __restrict__ A, const _Float16* __restrict__ W,
                                  int K, int m0, int n0, int kb, int tid,
                                  _Float16* la, _Float16* lb) {
  #pragma unroll
  for (int it = 0; it < 4; ++it) {
    int idx = tid + it * 256;            // 1024 uint4 per matrix
    int r = idx >> 3, w = idx & 7;
    cp16(&A[(size_t)(m0 + r) * K + kb + w * 8], &la[r * LSTR + w * 8]);
    cp16(&W[(size_t)(n0 + r) * K + kb + w * 8], &lb[r * LSTR + w * 8]);
  }
}

__device__ inline void stage_sim(const _Float16* __restrict__ n1, const _Float16* __restrict__ n2,
                                 int K, int i0, int jb, int kb, int tid,
                                 _Float16* la, _Float16* lb) {
  #pragma unroll
  for (int it = 0; it < 4; ++it) {
    int idx = tid + it * 256;
    int r = idx >> 3, w = idx & 7;
    const _Float16* srcA = (r < 64) ? &n1[(size_t)(i0 + r) * K] : &n2[(size_t)(i0 + r - 64) * K];
    cp16(&srcA[kb + w * 8], &la[r * LSTR + w * 8]);
    const _Float16* srcB = (r < 64) ? &n1[(size_t)(jb + r) * K] : &n2[(size_t)(jb + r - 64) * K];
    cp16(&srcB[kb + w * 8], &lb[r * LSTR + w * 8]);
  }
}

// ---------- NT GEMM: C[m,n] = act( sum_k A[m,k]*W[n,k] + bias[n] ) ----------
// MODE 0: ELU, f16 out (layer 1).  MODE 1: identity, f32 out (layer 2).
// min-blocks=1: occupancy is LDS-bound (72KB) anyway; let RA keep all fragments live.
template<int MODE>
__global__ __launch_bounds__(256, 1) void gemm_nt_kernel(
    const _Float16* __restrict__ A,   // [M,K] f16, row-major
    const _Float16* __restrict__ W,   // [Nn,K] f16, row-major (NT operand)
    const float*    __restrict__ bias,
    _Float16* __restrict__ outh, float* __restrict__ outf,
    int M, int Nn, int K)
{
  __shared__ _Float16 lA[2][128 * LSTR];
  __shared__ _Float16 lB[2][128 * LSTR];
  const int tid  = threadIdx.x;
  const int lane = tid & 31;
  const int wave = tid >> 5;      // 0..7 -> row strip
  const int g    = lane >> 4;     // half-wave group
  const int ln   = lane & 15;
  const int m0   = blockIdx.x * 128;
  const int n0   = blockIdx.y * 128;

  v8f acc[8];
  const v8f vzero = {0.f,0.f,0.f,0.f,0.f,0.f,0.f,0.f};
  #pragma unroll
  for (int i = 0; i < 8; ++i) acc[i] = vzero;

  int cur = 0;
  stage_gemm(A, W, K, m0, n0, 0, tid, lA[0], lB[0]);
  wait_async_lds();
  __syncthreads();

  for (int kb = 0; kb < K; kb += KC) {
    // software pipeline: issue async stage of next chunk into the other buffer
    if (kb + KC < K)
      stage_gemm(A, W, K, m0, n0, kb + KC, tid, lA[cur ^ 1], lB[cur ^ 1]);

    const _Float16* la = lA[cur];
    const _Float16* lb = lB[cur];
    #pragma unroll
    for (int ks = 0; ks < KC; ks += 32) {
      v16h a = ld_frag_a(&la[(wave * 16 + ln) * LSTR], ks + g * 16);
      v16h bf[8];
      #pragma unroll
      for (int ct = 0; ct < 8; ++ct)
        bf[ct] = ld_frag_b(&lb[(ct * 16 + ln) * LSTR], ks + g * 16);
      #pragma unroll
      for (int ct = 0; ct < 8; ++ct)
        acc[ct] = wmma_f16(a, bf[ct], acc[ct]);
    }
    wait_async_lds();
    __syncthreads();
    cur ^= 1;
  }

  // epilogue: C layout: VGPR r, lanes 0-15 -> row r, lanes 16-31 -> row r+8, col = ln
  #pragma unroll
  for (int ct = 0; ct < 8; ++ct) {
    int col = n0 + ct * 16 + ln;
    float bs = bias[col];
    #pragma unroll
    for (int r = 0; r < 8; ++r) {
      int row = m0 + wave * 16 + r + 8 * g;
      float v = acc[ct][r] + bs;
      if (MODE == 0) {
        v = v > 0.0f ? v : (__expf(v) - 1.0f);   // ELU
        outh[(size_t)row * Nn + col] = (_Float16)v;
      } else {
        outf[(size_t)row * Nn + col] = v;
      }
    }
  }
}

// ---------- row normalize + exact fp32 cross-dot ----------
__global__ __launch_bounds__(256) void normalize_kernel(
    const float* __restrict__ p1, const float* __restrict__ p2,
    _Float16* __restrict__ n1, _Float16* __restrict__ n2,
    float* __restrict__ dvec, int K)
{
  const int row = blockIdx.x;
  const int tid = threadIdx.x;
  const float* r1 = &p1[(size_t)row * K];
  const float* r2 = &p2[(size_t)row * K];
  float s1 = 0.f, s2 = 0.f, s12 = 0.f;
  for (int k = tid; k < K; k += 256) {
    float a = r1[k], b = r2[k];
    s1 += a * a; s2 += b * b; s12 += a * b;
  }
  __shared__ float sh1[256], sh2[256], sh3[256];
  sh1[tid] = s1; sh2[tid] = s2; sh3[tid] = s12;
  __syncthreads();
  for (int off = 128; off > 0; off >>= 1) {
    if (tid < off) { sh1[tid] += sh1[tid+off]; sh2[tid] += sh2[tid+off]; sh3[tid] += sh3[tid+off]; }
    __syncthreads();
  }
  float inv1 = 1.0f / fmaxf(sqrtf(sh1[0]), 1e-12f);
  float inv2 = 1.0f / fmaxf(sqrtf(sh2[0]), 1e-12f);
  for (int k = tid; k < K; k += 256) {
    n1[(size_t)row * K + k] = (_Float16)(r1[k] * inv1);
    n2[(size_t)row * K + k] = (_Float16)(r2[k] * inv2);
  }
  if (tid == 0) dvec[row] = sh3[0] * inv1 * inv2;
}

// ---------- fused similarity: S_ab[i] = sum_j exp(2 * n_a[i].n_b[j]) ----------
// Block: 64 i-rows of BOTH n1,n2. Waves 0-3: A=n1 strips, waves 4-7: A=n2 strips.
// j loop over 64-col tiles of both n1 and n2, full-K WMMA accumulation, exp+rowsum epilogue.
__global__ __launch_bounds__(256, 1) void sim_kernel(
    const _Float16* __restrict__ n1, const _Float16* __restrict__ n2,
    float* __restrict__ S11, float* __restrict__ S12,
    float* __restrict__ S21, float* __restrict__ S22,
    int Nr, int K)
{
  __shared__ _Float16 lA[2][128 * LSTR];   // rows 0-63: n1[i-block], 64-127: n2[i-block]
  __shared__ _Float16 lB[2][128 * LSTR];   // rows 0-63: n1[j-block], 64-127: n2[j-block]
  const int tid  = threadIdx.x;
  const int lane = tid & 31;
  const int wave = tid >> 5;
  const int g    = lane >> 4;
  const int ln   = lane & 15;
  const int i0   = blockIdx.x * 64;
  const int strip = wave & 3;
  const bool useN2 = wave >= 4;
  const int jspan  = Nr / gridDim.y;
  const int jstart = blockIdx.y * jspan;
  const int jend   = jstart + jspan;
  const v8f vzero = {0.f,0.f,0.f,0.f,0.f,0.f,0.f,0.f};

  float part1[8], part2[8];
  #pragma unroll
  for (int r = 0; r < 8; ++r) { part1[r] = 0.f; part2[r] = 0.f; }

  int cur = 0;
  stage_sim(n1, n2, K, i0, jstart, 0, tid, lA[0], lB[0]);
  wait_async_lds();
  __syncthreads();

  for (int jb = jstart; jb < jend; jb += 64) {
    v8f acc1[4], acc2[4];
    #pragma unroll
    for (int c = 0; c < 4; ++c) { acc1[c] = vzero; acc2[c] = vzero; }

    for (int kb = 0; kb < K; kb += KC) {
      // prefetch next (jb,kb) chunk into the other buffer
      int njb = jb, nkb = kb + KC;
      if (nkb == K) { njb = jb + 64; nkb = 0; }
      if (njb < jend)
        stage_sim(n1, n2, K, i0, njb, nkb, tid, lA[cur ^ 1], lB[cur ^ 1]);

      const _Float16* la = lA[cur];
      const _Float16* lb = lB[cur];
      #pragma unroll
      for (int ks = 0; ks < KC; ks += 32) {
        v16h a = ld_frag_a(&la[((useN2 ? 64 : 0) + strip * 16 + ln) * LSTR], ks + g * 16);
        v16h bf1[4], bf2[4];
        #pragma unroll
        for (int ct = 0; ct < 4; ++ct) {
          bf1[ct] = ld_frag_b(&lb[(ct * 16 + ln) * LSTR],        ks + g * 16);
          bf2[ct] = ld_frag_b(&lb[((64 + ct * 16) + ln) * LSTR], ks + g * 16);
        }
        #pragma unroll
        for (int ct = 0; ct < 4; ++ct) {
          acc1[ct] = wmma_f16(a, bf1[ct], acc1[ct]);
          acc2[ct] = wmma_f16(a, bf2[ct], acc2[ct]);
        }
      }
      wait_async_lds();
      __syncthreads();
      cur ^= 1;
    }
    // exp(logit/tau), tau = 0.5 -> exp(2x); accumulate per-lane row partials
    #pragma unroll
    for (int ct = 0; ct < 4; ++ct)
      #pragma unroll
      for (int r = 0; r < 8; ++r) {
        part1[r] += __expf(2.0f * acc1[ct][r]);
        part2[r] += __expf(2.0f * acc2[ct][r]);
      }
  }

  // reduce across the 16 lanes of each half-wave (cols), then atomically add row sums
  float* Sa = useN2 ? S21 : S11;   // vs n1 columns
  float* Sb = useN2 ? S22 : S12;   // vs n2 columns
  #pragma unroll
  for (int r = 0; r < 8; ++r) {
    float v1 = part1[r], v2 = part2[r];
    #pragma unroll
    for (int m = 1; m < 16; m <<= 1) {
      v1 += __shfl_xor(v1, m, 32);
      v2 += __shfl_xor(v2, m, 32);
    }
    if (ln == 0) {
      int row = i0 + strip * 16 + r + 8 * g;
      atomicAdd(&Sa[row], v1);
      atomicAdd(&Sb[row], v2);
    }
  }
}

// ---------- final loss: 0.5*(log(D1)+log(D2)) - d/tau ----------
__global__ void final_loss_kernel(
    const float* __restrict__ S11, const float* __restrict__ S12,
    const float* __restrict__ S21, const float* __restrict__ S22,
    const float* __restrict__ dvec, float* __restrict__ out, int n)
{
  int i = blockIdx.x * blockDim.x + threadIdx.x;
  if (i < n) {
    const float E2 = 7.38905609893065f;      // exp(1/tau) = exp(2): diag(refl)
    float D1 = S11[i] + S12[i] - E2;
    float D2 = S22[i] + S21[i] - E2;
    out[i] = 0.5f * (__logf(D1) + __logf(D2)) - 2.0f * dvec[i];
  }
}

extern "C" void kernel_launch(void* const* d_in, const int* in_sizes, int n_in,
                              void* d_out, int out_size, void* d_ws, size_t ws_size,
                              hipStream_t stream) {
  const int N = NROWS, D = DDIM;
  const float* z1  = (const float*)d_in[0];
  const float* z2  = (const float*)d_in[1];
  const float* w1  = (const float*)d_in[2];
  const float* b1  = (const float*)d_in[3];
  const float* w2  = (const float*)d_in[4];
  const float* b2  = (const float*)d_in[5];
  float* out = (float*)d_out;

  size_t off = 0;
  auto alloc = [&](size_t bytes) {
    void* p = (char*)d_ws + off;
    off += (bytes + 255) & ~(size_t)255;
    return p;
  };
  _Float16* z1h = (_Float16*)alloc((size_t)N * D * 2);
  _Float16* z2h = (_Float16*)alloc((size_t)N * D * 2);
  _Float16* w1h = (_Float16*)alloc((size_t)D * D * 2);
  _Float16* w2h = (_Float16*)alloc((size_t)D * D * 2);
  _Float16* h1h = (_Float16*)alloc((size_t)N * D * 2);
  _Float16* h2h = (_Float16*)alloc((size_t)N * D * 2);
  float*    p1  = (float*)alloc((size_t)N * D * 4);
  float*    p2  = (float*)alloc((size_t)N * D * 4);
  _Float16* n1h = (_Float16*)alloc((size_t)N * D * 2);
  _Float16* n2h = (_Float16*)alloc((size_t)N * D * 2);
  float*    dvec = (float*)alloc((size_t)N * 4);
  float*    S11  = (float*)alloc((size_t)N * 4);
  float*    S12  = (float*)alloc((size_t)N * 4);
  float*    S21  = (float*)alloc((size_t)N * 4);
  float*    S22  = (float*)alloc((size_t)N * 4);

  // 1) fp32 -> f16 conversions
  int nzd = N * D, nww = D * D;
  cvt_f16_kernel<<<(nzd + 255) / 256, 256, 0, stream>>>(z1, z1h, nzd);
  cvt_f16_kernel<<<(nzd + 255) / 256, 256, 0, stream>>>(z2, z2h, nzd);
  cvt_f16_kernel<<<(nww + 255) / 256, 256, 0, stream>>>(w1, w1h, nww);
  cvt_f16_kernel<<<(nww + 255) / 256, 256, 0, stream>>>(w2, w2h, nww);

  // 2) projection layer 1: h = ELU(z @ W1^T + b1)  (f16 out)
  dim3 g1(N / 128, D / 128);
  gemm_nt_kernel<0><<<g1, 256, 0, stream>>>(z1h, w1h, b1, h1h, nullptr, N, D, D);
  gemm_nt_kernel<0><<<g1, 256, 0, stream>>>(z2h, w1h, b1, h2h, nullptr, N, D, D);

  // 3) projection layer 2: p = h @ W2^T + b2  (f32 out for exact normalization)
  gemm_nt_kernel<1><<<g1, 256, 0, stream>>>(h1h, w2h, b2, nullptr, p1, N, D, D);
  gemm_nt_kernel<1><<<g1, 256, 0, stream>>>(h2h, w2h, b2, nullptr, p2, N, D, D);

  // 4) row-normalize + exact positive-pair dot d_i = n1_i . n2_i
  normalize_kernel<<<N, 256, 0, stream>>>(p1, p2, n1h, n2h, dvec, D);

  // 5) fused similarity row-sums (S arrays must start at zero; atomicAdd accumulation)
  zero_f32_kernel<<<(4 * N + 255) / 256, 256, 0, stream>>>(S11, 4 * N); // S11..S22 contiguous
  sim_kernel<<<dim3(N / 64, 4), 256, 0, stream>>>(n1h, n2h, S11, S12, S21, S22, N, D);

  // 6) final per-row loss
  final_loss_kernel<<<(N + 255) / 256, 256, 0, stream>>>(S11, S12, S21, S22, dvec, out, N);
}
